// BitLinear_6734508720681
// MI455X (gfx1250) — 8-bit matrix operand order repaired and checked by reading, compiled, not run on hardware
//
#include <hip/hip_runtime.h>
#include <stdint.h>

typedef __attribute__((ext_vector_type(8))) int v8i;

#define K_DIM 2048
#define N_DIM 2048
#define M_DIM 16384
#define W_ELEMS (2048 * 2048)

static constexpr size_t P_OFF   = 0;
static constexpr size_t WSC_OFF = 4096;
static constexpr size_t TS_OFF  = 8192;
static constexpr size_t XQ_OFF  = 131072;
static constexpr size_t WQ_OFF  = XQ_OFF + (size_t)M_DIM * K_DIM;

__global__ void k_wabs_partial(const float* __restrict__ w, float* __restrict__ partials) {
    __shared__ float red[256];
    const int tid = threadIdx.x;
    float s = 0.0f;
    for (size_t i = (size_t)blockIdx.x * 256 + tid; i < (size_t)W_ELEMS; i += (size_t)512 * 256)
        s += fabsf(w[i]);
    red[tid] = s;
    __syncthreads();
    for (int off = 128; off > 0; off >>= 1) {
        if (tid < off) red[tid] += red[tid + off];
        __syncthreads();
    }
    if (tid == 0) partials[blockIdx.x] = red[0];
}

__global__ void k_wscale_final(const float* __restrict__ partials, float* __restrict__ wscale) {
    __shared__ float red[256];
    const int tid = threadIdx.x;
    red[tid] = partials[tid] + partials[tid + 256];
    __syncthreads();
    for (int off = 128; off > 0; off >>= 1) {
        if (tid < off) red[tid] += red[tid + off];
        __syncthreads();
    }
    if (tid == 0) wscale[0] = red[0] / (float)W_ELEMS;
}

__global__ void k_wquant(const float* __restrict__ w, const float* __restrict__ wscale_p,
                         int8_t* __restrict__ wq) {
    const float t = 0.5f * wscale_p[0];
    const size_t base = ((size_t)blockIdx.x * 256 + threadIdx.x) * 8;
    float4 a = *(const float4*)(w + base);
    float4 b = *(const float4*)(w + base + 4);
    float v[8] = {a.x, a.y, a.z, a.w, b.x, b.y, b.z, b.w};
    uint64_t p = 0;
#pragma unroll
    for (int i = 0; i < 8; i++) {
        int8_t q = (v[i] > t) ? (int8_t)1 : ((v[i] < -t) ? (int8_t)-1 : (int8_t)0);
        p |= (uint64_t)(uint8_t)q << (8 * i);
    }
    *(uint64_t*)(wq + base) = p;
}

__global__ void k_actquant(const float* __restrict__ x, const float* __restrict__ wscale_p,
                           int8_t* __restrict__ xq, float* __restrict__ tscale) {
    __shared__ float red[256];
    const int tid = threadIdx.x;
    const size_t token = blockIdx.x;
    const float* row = x + token * (size_t)K_DIM;
    float4 a = *(const float4*)(row + tid * 8);
    float4 b = *(const float4*)(row + tid * 8 + 4);
    float v[8] = {a.x, a.y, a.z, a.w, b.x, b.y, b.z, b.w};
    float m = 0.0f;
#pragma unroll
    for (int i = 0; i < 8; i++) m = fmaxf(m, fabsf(v[i]));
    red[tid] = m;
    __syncthreads();
    for (int off = 128; off > 0; off >>= 1) {
        if (tid < off) red[tid] = fmaxf(red[tid], red[tid + off]);
        __syncthreads();
    }
    const float xs = fmaxf(red[0], 1e-5f);
    const float inv = 127.0f / xs;
    uint64_t p = 0;
#pragma unroll
    for (int i = 0; i < 8; i++) {
        float q = rintf(v[i] * inv);
        q = fminf(127.0f, fmaxf(-127.0f, q));
        p |= (uint64_t)(uint8_t)(int8_t)(int)q << (8 * i);
    }
    *(uint64_t*)(xq + token * (size_t)K_DIM + tid * 8) = p;
    if (tid == 0) tscale[token] = wscale_p[0] / xs;
}

__global__ __launch_bounds__(256) void k_gemm(const int8_t* __restrict__ xq,
                                              const int8_t* __restrict__ wq,
                                              const float* __restrict__ tscale,
                                              float* __restrict__ out) {
    const int lane = threadIdx.x & 31;
    const int wave = threadIdx.x >> 5;
    const int wm = wave & 3;
    const int wn = wave >> 2;
    const int l15 = lane & 15;
    const int hi = lane >> 4;

    const int rowBase = blockIdx.y * 128 + wm * 32;
    const int colBase = blockIdx.x * 128 + wn * 64;

    const int8_t* aBase = xq + (size_t)(rowBase + l15) * K_DIM + hi * 16;
    const int8_t* bBase = wq + (size_t)(colBase + l15) * K_DIM + hi * 16;

    auto loadA = [&](int k0, v8i a[2]) {
#pragma unroll
        for (int mi = 0; mi < 2; mi++) {
            const int8_t* p = aBase + (size_t)mi * 16 * K_DIM + k0;
            int2 d0 = *(const int2*)(p);
            int2 d1 = *(const int2*)(p + 8);
            int2 d2 = *(const int2*)(p + 32);
            int2 d3 = *(const int2*)(p + 40);
            v8i t;
            t[0] = d0.x; t[1] = d0.y; t[2] = d1.x; t[3] = d1.y;
            t[4] = d2.x; t[5] = d2.y; t[6] = d3.x; t[7] = d3.y;
            a[mi] = t;
        }
    };
    auto loadB = [&](int k0, v8i b[4]) {
#pragma unroll
        for (int ni = 0; ni < 4; ni++) {
            const int8_t* p = bBase + (size_t)ni * 16 * K_DIM + k0;
            int4 q0 = *(const int4*)(p);
            int4 q1 = *(const int4*)(p + 32);
            v8i t;
            t[0] = q0.x; t[1] = q0.y; t[2] = q0.z; t[3] = q0.w;
            t[4] = q1.x; t[5] = q1.y; t[6] = q1.z; t[7] = q1.w;
            b[ni] = t;
        }
    };

    v8i acc[2][4];
#pragma unroll
    for (int i = 0; i < 2; i++)
#pragma unroll
        for (int j = 0; j < 4; j++) acc[i][j] = {};

    v8i aCur[2], bCur[4];
    loadA(0, aCur);
    loadB(0, bCur);

#pragma unroll
    for (int k0 = 0; k0 < K_DIM; k0 += 64) {
        v8i aNxt[2], bNxt[4];
        if (k0 + 64 < K_DIM) {
            loadA(k0 + 64, aNxt);
            loadB(k0 + 64, bNxt);
        }
#pragma unroll
        for (int ni = 0; ni < 4; ni++)
#pragma unroll
            for (int mi = 0; mi < 2; mi++)
                acc[mi][ni] = __builtin_amdgcn_wmma_i32_16x16x64_iu8(
                     true, aCur[mi],  true, bCur[ni], acc[mi][ni],
                     false,  false);
        if (k0 + 64 < K_DIM) {
#pragma unroll
            for (int mi = 0; mi < 2; mi++) aCur[mi] = aNxt[mi];
#pragma unroll
            for (int ni = 0; ni < 4; ni++) bCur[ni] = bNxt[ni];
        }
    }

#pragma unroll
    for (int mi = 0; mi < 2; mi++) {
        const int rtile = rowBase + mi * 16 + hi * 8;
#pragma unroll
        for (int j = 0; j < 8; j++) {
            const int row = rtile + j;
            const float s = tscale[row];
#pragma unroll
            for (int ni = 0; ni < 4; ni++) {
                const int col = colBase + ni * 16 + l15;
                float v = (float)acc[mi][ni][j] * s;
                v = fmaxf(v, 0.0f);
                out[(size_t)row * N_DIM + col] = v * v;
            }
        }
    }
}

extern "C" void kernel_launch(void* const* d_in, const int* in_sizes, int n_in,
                              void* d_out, int out_size, void* d_ws, size_t ws_size,
                              hipStream_t stream) {
    const float* x = (const float*)d_in[0];
    const float* w = (const float*)d_in[1];
    float* out = (float*)d_out;

    char* ws = (char*)d_ws;
    float* partials = (float*)(ws + P_OFF);
    float* wscale   = (float*)(ws + WSC_OFF);
    float* tscale   = (float*)(ws + TS_OFF);
    int8_t* xq      = (int8_t*)(ws + XQ_OFF);
    int8_t* wq      = (int8_t*)(ws + WQ_OFF);

    k_wabs_partial<<<512, 256, 0, stream>>>(w, partials);
    k_wscale_final<<<1, 256, 0, stream>>>(partials, wscale);
    k_wquant<<<W_ELEMS / (256 * 8), 256, 0, stream>>>(w, wscale, wq);
    k_actquant<<<M_DIM, 256, 0, stream>>>(x, wscale, xq, tscale);

    dim3 grid(N_DIM / 128, M_DIM / 128);
    k_gemm<<<grid, 256, 0, stream>>>(xq, wq, tscale, out);
}
